// GCNSkeletonTokenizer_75849122447658
// MI455X (gfx1250) — matrix operand order repaired and checked by reading, not compiled, not run on hardware
//
#include <hip/hip_runtime.h>
#include <hip/hip_bf16.h>
#include <cstring>

typedef __attribute__((ext_vector_type(16))) _Float16 v16h;
typedef __attribute__((ext_vector_type(8)))  _Float16 v8h;
typedef __attribute__((ext_vector_type(8)))  float    v8f;

#define BN_SCALE_F 0.9999950000374997f  // 1/sqrt(1+1e-5)

struct AggDesc {
  long long vbase[6];
  float adj[6][6];
  int n;
};
struct ResDesc {
  long long rvbase[6];
};

// ---------------------------------------------------------------------------
// WMMA fragment index map (wave32; this file's contiguous k order, shared by A and B).
// Flat fragment index i = ((mt*(K/32)+kt)*32 + lane)*16 + e  ->  (m, k)
//   m = mt*16 + (lane&15)
//   k = kt*32 + 16*(lane>>4) + e
// Weights are pre-shuffled into this order by the prep kernel so an
// A-fragment load is one contiguous 32B per lane (coalesced b128 pair).
// ---------------------------------------------------------------------------
__host__ __device__ inline void frag_mk(int i, int K, int& m, int& k) {
  const int e = i & 15;
  const int lane = (i >> 4) & 31;
  const int tile = i >> 9;  // 32 lanes * 16 halves = 512 elems per 16x32 tile
  const int KT = K >> 5;
  const int mt = tile / KT;
  const int kt = tile - mt * KT;
  const int j = e >> 1;
  m = mt * 16 + (lane & 15);
  k = kt * 32 + 16 * (lane >> 4) + e;
}

// ---------------------------------------------------------------------------
// Embed: h[v][b*T+t][c] = x[b,t,v,:] . ew[c,:] + eb[c]   (fp16 out)
// ---------------------------------------------------------------------------
__global__ void embed_kernel(const float* __restrict__ x, const float* __restrict__ ew,
                             const float* __restrict__ eb, _Float16* __restrict__ h,
                             int BT) {
  const int gid = blockIdx.x * blockDim.x + threadIdx.x;
  const int total = BT * 25 * 64;
  if (gid >= total) return;
  const int c  = gid & 63;
  const int pv = gid >> 6;
  const int v  = pv % 25;
  const int bt = pv / 25;
  const float* xr = x + (size_t)(bt * 25 + v) * 3;
  float acc = eb[c] + xr[0] * ew[c * 3 + 0] + xr[1] * ew[c * 3 + 1] + xr[2] * ew[c * 3 + 2];
  h[((size_t)v * BT + bt) * 64 + c] = (_Float16)acc;
}

// ---------------------------------------------------------------------------
// Weight prep: collapse 3-partition sgcn weights; emit fp16 in WMMA fragment
// order. Per-group region layout (bytes):
//  0      Weff1 (64x64)          8192   tw1h (3x64x64)
//  32768  rw2h (128x64)          49152  Weff2 (128x64)
//  65536  tw2h (3x128x128)       163840 rw3h (64x128)
//  180224 Weff3 (64x128)         196608 tw3h (3x64x64)
//  221184 beff1 f32[64]  221440 beff2 f32[128]  221952 beff3 f32[64]
// ---------------------------------------------------------------------------
__global__ void prep_weights_kernel(
    const float* __restrict__ sw1, const float* __restrict__ sb1, const float* __restrict__ tw1,
    const float* __restrict__ rw2, const float* __restrict__ sw2, const float* __restrict__ tw2,
    const float* __restrict__ rw3, const float* __restrict__ sw3, const float* __restrict__ tw3,
    const float* __restrict__ sb2, const float* __restrict__ sb3,
    char* wbase, unsigned long long gstride) {
  const int g = blockIdx.x;
  const int tid = threadIdx.x;
  char* wb = wbase + (unsigned long long)g * gstride;
  _Float16* Weff1 = (_Float16*)(wb + 0);
  _Float16* tw1h  = (_Float16*)(wb + 8192);
  _Float16* rw2h  = (_Float16*)(wb + 32768);
  _Float16* Weff2 = (_Float16*)(wb + 49152);
  _Float16* tw2h  = (_Float16*)(wb + 65536);
  _Float16* rw3h  = (_Float16*)(wb + 163840);
  _Float16* Weff3 = (_Float16*)(wb + 180224);
  _Float16* tw3h  = (_Float16*)(wb + 196608);
  float* beff1 = (float*)(wb + 221184);
  float* beff2 = (float*)(wb + 221440);
  float* beff3 = (float*)(wb + 221952);
  int m, k;

  for (int i = tid; i < 64 * 64; i += blockDim.x) {
    frag_mk(i, 64, m, k);
    float s = sw1[(g * 192 + m) * 64 + k] + sw1[(g * 192 + 64 + m) * 64 + k] +
              sw1[(g * 192 + 128 + m) * 64 + k];
    Weff1[i] = (_Float16)s;
  }
  for (int i = tid; i < 64; i += blockDim.x)
    beff1[i] = sb1[g * 192 + i] + sb1[g * 192 + 64 + i] + sb1[g * 192 + 128 + i];
  for (int i = tid; i < 3 * 64 * 64; i += blockDim.x) {
    int tap = i >> 12, ii = i & 4095;
    frag_mk(ii, 64, m, k);
    tw1h[i] = (_Float16)tw1[(((g * 64 + m) * 64 + k) * 3) + tap];
  }
  for (int i = tid; i < 128 * 64; i += blockDim.x) {
    frag_mk(i, 64, m, k);
    rw2h[i] = (_Float16)rw2[((size_t)g * 128 + m) * 64 + k];
  }
  for (int i = tid; i < 128 * 64; i += blockDim.x) {
    frag_mk(i, 64, m, k);
    float s = sw2[(g * 384 + m) * 64 + k] + sw2[(g * 384 + 128 + m) * 64 + k] +
              sw2[(g * 384 + 256 + m) * 64 + k];
    Weff2[i] = (_Float16)s;
  }
  for (int i = tid; i < 128; i += blockDim.x)
    beff2[i] = sb2[g * 384 + i] + sb2[g * 384 + 128 + i] + sb2[g * 384 + 256 + i];
  for (int i = tid; i < 3 * 128 * 128; i += blockDim.x) {
    int tap = i >> 14, ii = i & 16383;
    frag_mk(ii, 128, m, k);
    tw2h[i] = (_Float16)tw2[(((size_t)(g * 128 + m) * 128 + k) * 3) + tap];
  }
  for (int i = tid; i < 64 * 128; i += blockDim.x) {
    frag_mk(i, 128, m, k);
    rw3h[i] = (_Float16)rw3[((size_t)g * 64 + m) * 128 + k];
  }
  for (int i = tid; i < 64 * 128; i += blockDim.x) {
    frag_mk(i, 128, m, k);
    float s = sw3[(g * 192 + m) * 128 + k] + sw3[(g * 192 + 64 + m) * 128 + k] +
              sw3[(g * 192 + 128 + m) * 128 + k];
    Weff3[i] = (_Float16)s;
  }
  for (int i = tid; i < 64; i += blockDim.x)
    beff3[i] = sb3[g * 192 + i] + sb3[g * 192 + 64 + i] + sb3[g * 192 + 128 + i];
  for (int i = tid; i < 3 * 64 * 64; i += blockDim.x) {
    int tap = i >> 12, ii = i & 4095;
    frag_mk(ii, 64, m, k);
    tw3h[i] = (_Float16)tw3[(((g * 64 + m) * 64 + k) * 3) + tap];
  }
}

// ---------------------------------------------------------------------------
// sgcn: adjacency-aggregate (or identity) + channel GEMM via WMMA.
// LDS tile is position-major with K contiguous so a B-fragment is one
// aligned 32B LDS read; A-fragments are contiguous 32B global reads.
// ---------------------------------------------------------------------------
template <int KIN, int O, bool BN>
__global__ __launch_bounds__(128) void sgcn_wmma_kernel(
    const _Float16* __restrict__ in, _Float16* __restrict__ out,
    const _Float16* __restrict__ W, const float* __restrict__ bias,
    const float* __restrict__ bng, const float* __restrict__ bnb,
    int B, int T, AggDesc gd) {
  constexpr int SROW = KIN + 16;  // halves; row stride multiple of 32B
  __shared__ alignas(32) _Float16 stage[4][16][SROW];
  const int lane = threadIdx.x & 31;
  const int wave = threadIdx.x >> 5;
  const int BT = B * T;
  const int p0 = (blockIdx.x * 4 + wave) * 16;
  const int u = p0 / BT;
  const int rem = p0 - u * BT;
  const int b = rem / T;
  const int t0 = rem - b * T;

  // stage: stage[n][k] = sum_v adj[u][v] * in[v][b,t0+n][k]  (8-ch chunks)
  for (int cid = lane; cid < 2 * KIN; cid += 32) {
    const int n2 = cid & 15;
    const int kc = cid >> 4;  // 8-half chunk index
    const long long boff = (long long)(b * T + t0 + n2) * KIN + kc * 8;
    float acc[8];
#pragma unroll
    for (int j = 0; j < 8; ++j) acc[j] = 0.f;
    for (int v = 0; v < gd.n; ++v) {
      const v8h xv = *(const v8h*)(in + gd.vbase[v] + boff);
      const float w = gd.adj[u][v];
#pragma unroll
      for (int j = 0; j < 8; ++j) acc[j] += w * (float)xv[j];
    }
    v8h o;
#pragma unroll
    for (int j = 0; j < 8; ++j) o[j] = (_Float16)acc[j];
    *(v8h*)(&stage[wave][n2][kc * 8]) = o;
  }
  __syncthreads();

  const int nn = lane & 15;
  const int hi = lane >> 4;
  const int mb = hi * 8;
#pragma unroll
  for (int mt = 0; mt < O / 16; ++mt) {
    const int m0 = mt * 16;
    v8f c = *(const v8f*)(bias + m0 + mb);
#pragma unroll
    for (int kt = 0; kt < KIN; kt += 32) {
      const int tile = mt * (KIN / 32) + (kt >> 5);
      const v16h a = *(const v16h*)(W + (((size_t)tile * 32 + lane) << 4));
      const v16h bf = *(const v16h*)(&stage[wave][nn][kt + 16 * hi]);
      c = __builtin_amdgcn_wmma_f32_16x16x32_f16(false, a, false, bf, (short)0, c,
                                                 false, false);
    }
    if constexpr (BN) {
      const v8f gv = *(const v8f*)(bng + m0 + mb);
      const v8f bv = *(const v8f*)(bnb + m0 + mb);
#pragma unroll
      for (int r = 0; r < 8; ++r) c[r] = c[r] * (gv[r] * BN_SCALE_F) + bv[r];
    }
    v8h oo;
#pragma unroll
    for (int r = 0; r < 8; ++r) oo[r] = (_Float16)c[r];
    *(v8h*)(out + (long long)(p0 + nn) * O + m0 + mb) = oo;
  }
}

// ---------------------------------------------------------------------------
// Temporal conv (k=3, pad 1) + residual + BN + ReLU, all via WMMA
// ---------------------------------------------------------------------------
template <int KIN, int O>
__global__ __launch_bounds__(128) void tconv_wmma_kernel(
    const _Float16* __restrict__ yin, const _Float16* __restrict__ resin,
    _Float16* __restrict__ out, const _Float16* __restrict__ Wt,
    const float* __restrict__ tb, const float* __restrict__ bng,
    const float* __restrict__ bnb, int B, int T, ResDesc rd) {
  constexpr int SROW = KIN + 16;
  __shared__ alignas(32) _Float16 stage[4][16][SROW];
  const int lane = threadIdx.x & 31;
  const int wave = threadIdx.x >> 5;
  const int BT = B * T;
  const int p0 = (blockIdx.x * 4 + wave) * 16;
  const int u = p0 / BT;
  const int rem = p0 - u * BT;
  const int b = rem / T;
  const int t0 = rem - b * T;
  const int nn = lane & 15;
  const int hi = lane >> 4;
  const int mb = hi * 8;

  v8f acc[O / 16];
#pragma unroll
  for (int mt = 0; mt < O / 16; ++mt)
    acc[mt] = *(const v8f*)(tb + mt * 16 + mb);

  for (int tap = 0; tap < 3; ++tap) {
    __syncthreads();
    for (int cid = lane; cid < 2 * KIN; cid += 32) {
      const int n2 = cid & 15;
      const int kc = cid >> 4;
      const int t = t0 + n2 + tap - 1;
      v8h o;
      if (t >= 0 && t < T) {
        o = *(const v8h*)(yin + (long long)((u * B + b) * T + t) * KIN + kc * 8);
      } else {
#pragma unroll
        for (int j = 0; j < 8; ++j) o[j] = (_Float16)0.f;
      }
      *(v8h*)(&stage[wave][n2][kc * 8]) = o;
    }
    __syncthreads();
    const _Float16* Wtap = Wt + (size_t)tap * O * KIN;
#pragma unroll
    for (int mt = 0; mt < O / 16; ++mt) {
#pragma unroll
      for (int kt = 0; kt < KIN; kt += 32) {
        const int tile = mt * (KIN / 32) + (kt >> 5);
        const v16h a = *(const v16h*)(Wtap + (((size_t)tile * 32 + lane) << 4));
        const v16h bf = *(const v16h*)(&stage[wave][nn][kt + 16 * hi]);
        acc[mt] = __builtin_amdgcn_wmma_f32_16x16x32_f16(false, a, false, bf, (short)0,
                                                         acc[mt], false, false);
      }
    }
  }
  // epilogue: + residual, BN, ReLU (vectorized, branchless)
  const long long btl = (long long)(b * T + t0 + nn);
  const long long prow = (long long)(p0 + nn) * O;
  const long long roff = rd.rvbase[u] + btl * O;
#pragma unroll
  for (int mt = 0; mt < O / 16; ++mt) {
    const int m0 = mt * 16;
    const v8h rres = *(const v8h*)(resin + roff + m0 + mb);
    const v8f gv = *(const v8f*)(bng + m0 + mb);
    const v8f bv = *(const v8f*)(bnb + m0 + mb);
    v8h oo;
#pragma unroll
    for (int r = 0; r < 8; ++r) {
      float v = acc[mt][r] + (float)rres[r];
      v = v * (gv[r] * BN_SCALE_F) + bv[r];
      v = v > 0.f ? v : 0.f;
      oo[r] = (_Float16)v;
    }
    *(v8h*)(out + prow + m0 + mb) = oo;
  }
}

// ---------------------------------------------------------------------------
// VQ: per (group,batch) block — mean over (T,V) with vector loads,
// 128 distances, argmin, quant/idx/loss outputs
// ---------------------------------------------------------------------------
__global__ void zero_loss_kernel(float* out, int idx) {
  if (threadIdx.x == 0 && blockIdx.x == 0) out[idx] = 0.f;
}

__global__ __launch_bounds__(128) void vq_kernel(const _Float16* __restrict__ act,
                                                 const float* __restrict__ cb,
                                                 float* __restrict__ dout, int g, int n,
                                                 int T, int B) {
  __shared__ float red[128][8];
  __shared__ float feat[64];
  __shared__ float dbuf[128];
  __shared__ int ibuf[128];
  __shared__ int bestIdx;
  const int b = blockIdx.x;
  const int tid = threadIdx.x;

  // partial sums: thread (pi,co) accumulates 8 channels over strided positions
  {
    const int co = tid & 7;
    const int pi = tid >> 3;
    float acc[8];
#pragma unroll
    for (int j = 0; j < 8; ++j) acc[j] = 0.f;
    for (int p = pi; p < n * T; p += 16) {
      const int u = p / T, t = p - u * T;
      const v8h xv =
          *(const v8h*)(act + ((size_t)(u * B + b) * T + t) * 64 + co * 8);
#pragma unroll
      for (int j = 0; j < 8; ++j) acc[j] += (float)xv[j];
    }
#pragma unroll
    for (int j = 0; j < 8; ++j) red[tid][j] = acc[j];
  }
  __syncthreads();
  if (tid < 64) {
    const int chunk = tid >> 3, jj = tid & 7;
    float s = 0.f;
#pragma unroll
    for (int pi = 0; pi < 16; ++pi) s += red[pi * 8 + chunk][jj];
    feat[tid] = s / (float)(n * T);
  }
  __syncthreads();
  {
    const float* crow = cb + ((size_t)g * 128 + tid) * 64;
    float d = 0.f;
    for (int o = 0; o < 64; ++o) {
      float df = feat[o] - crow[o];
      d += df * df;
    }
    dbuf[tid] = d;
    ibuf[tid] = tid;
  }
  __syncthreads();
  for (int s = 64; s > 0; s >>= 1) {
    if (tid < s) {
      if (dbuf[tid + s] < dbuf[tid] ||
          (dbuf[tid + s] == dbuf[tid] && ibuf[tid + s] < ibuf[tid])) {
        dbuf[tid] = dbuf[tid + s];
        ibuf[tid] = ibuf[tid + s];
      }
    }
    __syncthreads();
  }
  if (tid == 0) {
    bestIdx = ibuf[0];
    ((int*)dout)[5 * 64 * 64 + 1 + g * 64 + b] = ibuf[0];
    // loss = sum_g 1.25 * mean_{b,o}((q-f)^2); d_min == ||q-f||^2 over 64 ch
    atomicAdd(&dout[5 * 64 * 64], 1.25f * dbuf[0] / 4096.f);
  }
  __syncthreads();
  if (tid < 64)
    dout[(size_t)g * 64 * 64 + (size_t)b * 64 + tid] =
        cb[((size_t)g * 128 + bestIdx) * 64 + tid];
}

// ---------------------------------------------------------------------------
// Host
// ---------------------------------------------------------------------------
extern "C" void kernel_launch(void* const* d_in, const int* in_sizes, int n_in,
                              void* d_out, int out_size, void* d_ws, size_t ws_size,
                              hipStream_t stream) {
  (void)in_sizes; (void)n_in; (void)out_size; (void)ws_size;
  const int B = 64, T = 256, BT = 16384;

  const float* x    = (const float*)d_in[0];
  const float* ew   = (const float*)d_in[1];
  const float* eb   = (const float*)d_in[2];
  const float* sw1  = (const float*)d_in[3];
  const float* sb1  = (const float*)d_in[4];
  const float* tw1  = (const float*)d_in[5];
  const float* tb1  = (const float*)d_in[6];
  const float* bg1  = (const float*)d_in[7];
  const float* bb1  = (const float*)d_in[8];
  const float* rw2  = (const float*)d_in[9];
  const float* rb2  = (const float*)d_in[10];
  const float* rg2  = (const float*)d_in[11];
  const float* rbb2 = (const float*)d_in[12];
  const float* sw2  = (const float*)d_in[13];
  const float* sb2  = (const float*)d_in[14];
  const float* tw2  = (const float*)d_in[15];
  const float* tb2  = (const float*)d_in[16];
  const float* bg2  = (const float*)d_in[17];
  const float* bb2  = (const float*)d_in[18];
  const float* rw3  = (const float*)d_in[19];
  const float* rb3  = (const float*)d_in[20];
  const float* rg3  = (const float*)d_in[21];
  const float* rbb3 = (const float*)d_in[22];
  const float* sw3  = (const float*)d_in[23];
  const float* sb3  = (const float*)d_in[24];
  const float* tw3  = (const float*)d_in[25];
  const float* tb3  = (const float*)d_in[26];
  const float* bg3  = (const float*)d_in[27];
  const float* bb3  = (const float*)d_in[28];
  const float* cb   = (const float*)d_in[29];
  float* out = (float*)d_out;

  // workspace carve
  char* ws = (char*)d_ws;
  size_t off = 0;
  auto carve = [&](size_t bytes) {
    size_t o = off;
    off += (bytes + 255) & ~(size_t)255;
    return o;
  };
  const size_t h_off = carve((size_t)25 * BT * 64 * 2);
  const size_t A_off = carve((size_t)6 * BT * 128 * 2);
  const size_t Y_off = carve((size_t)6 * BT * 128 * 2);
  const size_t R_off = carve((size_t)6 * BT * 128 * 2);
  const size_t GSTRIDE = 229376;  // per-group weight region
  const size_t w_off = carve(GSTRIDE * 5);

  _Float16* h    = (_Float16*)(ws + h_off);
  _Float16* bufA = (_Float16*)(ws + A_off);
  _Float16* bufY = (_Float16*)(ws + Y_off);
  _Float16* bufR = (_Float16*)(ws + R_off);
  char* wbase    = ws + w_off;

  static const int G_N[5] = {5, 6, 6, 4, 4};
  static const int G_J[5][6] = {{0, 1, 2, 3, 20, 0},  {4, 5, 6, 7, 21, 22},
                                {8, 9, 10, 11, 23, 24}, {12, 13, 14, 15, 0, 0},
                                {16, 17, 18, 19, 0, 0}};

  // --- host adjacency (mirrors _full_adj / _sub_adj, float64 math) ---
  double fa[25][25] = {};
  static const int E[24][2] = {{3, 2},  {2, 20}, {20, 1}, {1, 0},  {20, 4}, {4, 5},
                               {5, 6},  {6, 22}, {6, 7},  {7, 21}, {20, 8}, {8, 9},
                               {9, 10}, {10, 24},{10, 11},{11, 23},{0, 12}, {12, 13},
                               {13, 14},{14, 15},{0, 16}, {16, 17},{17, 18},{18, 19}};
  for (int e = 0; e < 24; ++e) {
    int i = E[e][0] - 1, j = E[e][1] - 1;
    if (i >= 0 && i < 25 && j >= 0 && j < 25) { fa[i][j] = 1.0; fa[j][i] = 1.0; }
  }
  for (int i = 0; i < 25; ++i) fa[i][i] += 1.0;
  for (int i = 0; i < 25; ++i) {
    double deg = 0;
    for (int j = 0; j < 25; ++j) deg += fa[i][j];
    if (deg == 0) deg = 1.0;
    for (int j = 0; j < 25; ++j) fa[i][j] /= deg;
  }
  float subadj[5][6][6];
  for (int g = 0; g < 5; ++g) {
    const int n = G_N[g];
    double sa[6][6] = {};
    for (int i = 0; i < n; ++i)
      for (int j = 0; j < n; ++j) sa[i][j] = fa[G_J[g][i]][G_J[g][j]];
    if (n > 1) {
      for (int i = 0; i < n; ++i) sa[i][i] += 1.0;
      double degv[6];
      for (int i = 0; i < n; ++i) {
        double d = 0;
        for (int j = 0; j < n; ++j) d += sa[i][j];
        degv[i] = d;
      }
      for (int i = 0; i < n; ++i)
        if (degv[i] == 1.0 && i != 0) { sa[0][i] = 1.0; sa[i][0] = 1.0; }
    }
    for (int i = 0; i < n; ++i) {
      double deg = 0;
      for (int j = 0; j < n; ++j) deg += sa[i][j];
      if (deg == 0) deg = 1.0;
      for (int j = 0; j < n; ++j) subadj[g][i][j] = (float)(sa[i][j] / deg);
    }
  }

  // --- launches ---
  prep_weights_kernel<<<5, 256, 0, stream>>>(sw1, sb1, tw1, rw2, sw2, tw2, rw3, sw3,
                                             tw3, sb2, sb3, wbase,
                                             (unsigned long long)GSTRIDE);
  {
    const int total = BT * 25 * 64;
    embed_kernel<<<(total + 255) / 256, 256, 0, stream>>>(x, ew, eb, h, BT);
  }
  zero_loss_kernel<<<1, 32, 0, stream>>>(out, 5 * 64 * 64);

  for (int g = 0; g < 5; ++g) {
    const int n = G_N[g];
    const int NB = n * 256;  // (n*BT)/64 position tiles per 4-wave block
    char* wb = wbase + (size_t)g * GSTRIDE;
    const _Float16* Weff1 = (const _Float16*)(wb + 0);
    const _Float16* tw1h  = (const _Float16*)(wb + 8192);
    const _Float16* rw2h  = (const _Float16*)(wb + 32768);
    const _Float16* Weff2 = (const _Float16*)(wb + 49152);
    const _Float16* tw2h  = (const _Float16*)(wb + 65536);
    const _Float16* rw3h  = (const _Float16*)(wb + 163840);
    const _Float16* Weff3 = (const _Float16*)(wb + 180224);
    const _Float16* tw3h  = (const _Float16*)(wb + 196608);
    const float* beff1 = (const float*)(wb + 221184);
    const float* beff2 = (const float*)(wb + 221440);
    const float* beff3 = (const float*)(wb + 221952);

    AggDesc agH{};  // layer1: gather joints from h, real adjacency
    for (int v = 0; v < n; ++v) agH.vbase[v] = (long long)G_J[g][v] * BT * 64;
    for (int i = 0; i < n; ++i)
      for (int j = 0; j < n; ++j) agH.adj[i][j] = subadj[g][i][j];
    agH.n = n;

    AggDesc agA64{}, agI64{}, agA128{}, agI128{};
    for (int v = 0; v < n; ++v) {
      agA64.vbase[v] = (long long)v * BT * 64;
      agI64.vbase[v] = (long long)v * BT * 64;
      agA128.vbase[v] = (long long)v * BT * 128;
      agI128.vbase[v] = (long long)v * BT * 128;
    }
    for (int i = 0; i < n; ++i)
      for (int j = 0; j < n; ++j) {
        agA64.adj[i][j] = subadj[g][i][j];
        agA128.adj[i][j] = subadj[g][i][j];
        agI64.adj[i][j] = (i == j) ? 1.f : 0.f;
        agI128.adj[i][j] = (i == j) ? 1.f : 0.f;
      }
    agA64.n = agI64.n = agA128.n = agI128.n = n;

    // ---- Layer 1: sgcn -> tconv (+identity residual from h) ----
    sgcn_wmma_kernel<64, 64, false><<<NB, 128, 0, stream>>>(h, bufY, Weff1, beff1,
                                                            nullptr, nullptr, B, T, agH);
    ResDesc r1{};
    for (int v = 0; v < n; ++v) r1.rvbase[v] = (long long)G_J[g][v] * BT * 64;
    tconv_wmma_kernel<64, 64><<<NB, 128, 0, stream>>>(bufY, h, bufA, tw1h, tb1 + g * 64,
                                                      bg1 + g * 64, bb1 + g * 64, B, T,
                                                      r1);

    // ---- Layer 2: residual proj (BN), sgcn, tconv ----
    sgcn_wmma_kernel<64, 128, true><<<NB, 128, 0, stream>>>(
        bufA, bufR, rw2h, rb2 + g * 128, rg2 + g * 128, rbb2 + g * 128, B, T, agI64);
    sgcn_wmma_kernel<64, 128, false><<<NB, 128, 0, stream>>>(
        bufA, bufY, Weff2, beff2, nullptr, nullptr, B, T, agA64);
    ResDesc r2{};
    for (int v = 0; v < n; ++v) r2.rvbase[v] = (long long)v * BT * 128;
    tconv_wmma_kernel<128, 128><<<NB, 128, 0, stream>>>(bufY, bufR, bufA, tw2h,
                                                        tb2 + g * 128, bg2 + g * 128,
                                                        bb2 + g * 128, B, T, r2);

    // ---- Layer 3: residual proj (BN), sgcn, tconv ----
    sgcn_wmma_kernel<128, 64, true><<<NB, 128, 0, stream>>>(
        bufA, bufR, rw3h, rb3 + g * 64, rg3 + g * 64, rbb3 + g * 64, B, T, agI128);
    sgcn_wmma_kernel<128, 64, false><<<NB, 128, 0, stream>>>(
        bufA, bufY, Weff3, beff3, nullptr, nullptr, B, T, agA128);
    ResDesc r3{};
    for (int v = 0; v < n; ++v) r3.rvbase[v] = (long long)v * BT * 64;
    tconv_wmma_kernel<64, 64><<<NB, 128, 0, stream>>>(bufY, bufR, bufA, tw3h,
                                                      tb3 + g * 64, bg3 + g * 64,
                                                      bb3 + g * 64, B, T, r3);

    // ---- Mean pool + VQ for this group (bufA reused next group) ----
    vq_kernel<<<64, 128, 0, stream>>>(bufA, cb, out, g, n, T, B);
  }
}
